// EfficientSelfAttention_35261681500234
// MI455X (gfx1250) — matrix operand order repaired and checked by reading, not compiled, not run on hardware
//
#include <hip/hip_runtime.h>

// ---------------------------------------------------------------------------
// CDNA5 (gfx1250) wave32 WMMA implementation of spatial-reduction attention.
// All GEMMs use v_wmma_f32_16x16x32_bf16 (bf16 in, f32 accumulate).
// Attention is fused (flash-style) so the 512MB attn matrix never hits HBM.
// LDS staging uses GLOBAL_LOAD_ASYNC_TO_LDS_B128 (ASYNCcnt) when available.
// ---------------------------------------------------------------------------

typedef __attribute__((ext_vector_type(16))) __bf16 v16bf;
typedef __attribute__((ext_vector_type(8)))  float  v8f;
typedef __attribute__((ext_vector_type(4)))  int    v4i;

static constexpr int TB = 256;  // 8 waves per block (wave32)

#ifndef __has_builtin
#define __has_builtin(x) 0
#endif
#if __has_builtin(__builtin_amdgcn_global_load_async_to_lds_b128) && \
    __has_builtin(__builtin_amdgcn_s_wait_asynccnt)
#define HAVE_ASYNC_LDS 1
#else
#define HAVE_ASYNC_LDS 0
#endif

#if HAVE_ASYNC_LDS
typedef __attribute__((address_space(1))) v4i as1_v4i;  // global (AS1)
typedef __attribute__((address_space(3))) v4i as3_v4i;  // LDS    (AS3)
__device__ inline void async_cp128(const void* g, void* l) {
  __builtin_amdgcn_global_load_async_to_lds_b128((as1_v4i*)g, (as3_v4i*)l, 0, 0);
}
__device__ inline void async_wait0() { __builtin_amdgcn_s_wait_asynccnt(0); }
#endif

__device__ inline unsigned short f2bf(float f) {
  unsigned u = __float_as_uint(f);
  u += 0x7fffu + ((u >> 16) & 1u);       // round-to-nearest-even
  return (unsigned short)(u >> 16);
}

__device__ inline v16bf frag16(uint4 lo, uint4 hi) {
  union { uint4 q[2]; v16bf v; } u;
  u.q[0] = lo; u.q[1] = hi;
  return u.v;
}

// ---------------- f32 -> bf16 conversion ----------------
__global__ __launch_bounds__(TB) void cvt_bf16_kernel(const float* __restrict__ src,
                                                      unsigned short* __restrict__ dst,
                                                      int n) {
  int i = blockIdx.x * TB + threadIdx.x;
  if (i < n) dst[i] = f2bf(src[i]);
}

// f32 [K][N] -> bf16 [N][K]  (pre-transposed weights so GEMM B-staging is a
// straight 128-bit block copy, async-LDS eligible)
__global__ __launch_bounds__(TB) void cvtT_bf16_kernel(const float* __restrict__ src,
                                                       unsigned short* __restrict__ dst,
                                                       int K, int N) {
  int i = blockIdx.x * TB + threadIdx.x;   // destination-linear: i = n*K + k
  if (i < K * N) {
    int n = i / K, k = i - n * K;
    dst[i] = f2bf(src[(size_t)k * N + n]);
  }
}

// ---------------- im2row for the 2x2/stride-2 SR conv ----------------
// xp[b*1024 + i*32 + j][(di*2+dj)*512 + c] = x[b][2i+di][2j+dj][c]   (bf16)
__global__ __launch_bounds__(TB) void im2row_kernel(const float* __restrict__ x,
                                                    unsigned short* __restrict__ xp) {
  int vid = blockIdx.x * TB + threadIdx.x;   // 4096*2048/8 = 1,048,576 vec8s
  int row = vid >> 8;
  int kk  = (vid & 255) * 8;
  int grp = kk >> 9, c = kk & 511;
  int di = grp >> 1, dj = grp & 1;
  int b = row >> 10, ij = row & 1023;
  int i = ij >> 5, j = ij & 31;
  const float* s = x + (((size_t)((b * 64 + i * 2 + di) * 64 + j * 2 + dj)) << 9) + c;
  float4 a0 = *(const float4*)s;
  float4 a1 = *(const float4*)(s + 4);
  unsigned short t[8];
  t[0]=f2bf(a0.x); t[1]=f2bf(a0.y); t[2]=f2bf(a0.z); t[3]=f2bf(a0.w);
  t[4]=f2bf(a1.x); t[5]=f2bf(a1.y); t[6]=f2bf(a1.z); t[7]=f2bf(a1.w);
  *(uint4*)&xp[((size_t)row << 11) + kk] = *(const uint4*)t;
}

// ---------------- tiled bf16 GEMM: C = A[MxK] * Bt[NxK]^T + bias ----------------
// Bt is the PRE-TRANSPOSED weight ([N][K] row-major, K contiguous).
// mode 0: q-pack     -> obf0[((b*8+h)*4096+n)*64+d]         (bf16)
// mode 1: row-major  -> of32[row*N+col]                      (f32)
// mode 2: kv-pack    -> k: obf0[((b*8+h)*1024+n)*64+d]
//                       v: obf1[((b*8+h)*64+d)*1024+n]       (bf16, v transposed)
#define BMT 128
#define BNT 128
#define BKT 32

__global__ __launch_bounds__(TB) void gemm_bf16_kernel(
    const unsigned short* __restrict__ A, const unsigned short* __restrict__ Bt,
    const float* __restrict__ bias, int M, int N, int K, int mode,
    unsigned short* __restrict__ obf0, unsigned short* __restrict__ obf1,
    float* __restrict__ of32) {
  __shared__ __align__(16) unsigned short As[BMT * BKT];   // [m][k] row-major
  __shared__ __align__(16) unsigned short Bst[BNT * BKT];  // [n][k] row-major
  const int tid = threadIdx.x, lane = tid & 31, wave = tid >> 5;
  const int nbl = N / BNT;
  const int mb = (blockIdx.x / nbl) * BMT;
  const int nb = (blockIdx.x % nbl) * BNT;
  const int wm = (wave >> 2) * 64;   // 2 wave-rows of 64
  const int wn = (wave & 3) * 32;    // 4 wave-cols of 32
  const int r = lane & 15, kh = lane >> 4;

  // per-thread staging coordinates: two 8-element vectors per tile
  const int v0 = tid, v1 = tid + TB;                // 512 vec8 per 128x32 tile
  const int sm0 = v0 >> 2, sk0 = (v0 & 3) * 8;
  const int sm1 = v1 >> 2, sk1 = (v1 & 3) * 8;

  v8f zero = {};
  v8f acc[4][2];
#pragma unroll
  for (int mt = 0; mt < 4; ++mt)
#pragma unroll
    for (int nt = 0; nt < 2; ++nt) acc[mt][nt] = zero;

  for (int kb = 0; kb < K; kb += BKT) {
    __syncthreads();
#if HAVE_ASYNC_LDS
    async_cp128(&A [(size_t)(mb + sm0) * K + kb + sk0], &As [sm0 * BKT + sk0]);
    async_cp128(&A [(size_t)(mb + sm1) * K + kb + sk1], &As [sm1 * BKT + sk1]);
    async_cp128(&Bt[(size_t)(nb + sm0) * K + kb + sk0], &Bst[sm0 * BKT + sk0]);
    async_cp128(&Bt[(size_t)(nb + sm1) * K + kb + sk1], &Bst[sm1 * BKT + sk1]);
    async_wait0();
#else
    uint4 ta0 = *(const uint4*)&A [(size_t)(mb + sm0) * K + kb + sk0];
    uint4 ta1 = *(const uint4*)&A [(size_t)(mb + sm1) * K + kb + sk1];
    uint4 tb0 = *(const uint4*)&Bt[(size_t)(nb + sm0) * K + kb + sk0];
    uint4 tb1 = *(const uint4*)&Bt[(size_t)(nb + sm1) * K + kb + sk1];
    *(uint4*)&As [sm0 * BKT + sk0] = ta0;
    *(uint4*)&As [sm1 * BKT + sk1] = ta1;
    *(uint4*)&Bst[sm0 * BKT + sk0] = tb0;
    *(uint4*)&Bst[sm1 * BKT + sk1] = tb1;
#endif
    __syncthreads();

    v16bf af[4], bfr[2];
#pragma unroll
    for (int mt = 0; mt < 4; ++mt) {
      int row = wm + mt * 16 + r;
      uint4 lo = *(const uint4*)&As[row * BKT + kh * 8];
      uint4 hi = *(const uint4*)&As[row * BKT + 16 + kh * 8];
      af[mt] = frag16(lo, hi);
    }
#pragma unroll
    for (int nt = 0; nt < 2; ++nt) {
      int col = wn + nt * 16 + r;
      uint4 lo = *(const uint4*)&Bst[col * BKT + kh * 8];
      uint4 hi = *(const uint4*)&Bst[col * BKT + 16 + kh * 8];
      bfr[nt] = frag16(lo, hi);
    }
#pragma unroll
    for (int mt = 0; mt < 4; ++mt)
#pragma unroll
      for (int nt = 0; nt < 2; ++nt)
        acc[mt][nt] = __builtin_amdgcn_wmma_f32_16x16x32_bf16(
            false, af[mt], false, bfr[nt], (short)0, acc[mt][nt], false, false);
  }

  // epilogue (C layout: VGPR g -> row g + 8*(lane>=16), col lane&15)
#pragma unroll
  for (int mt = 0; mt < 4; ++mt)
#pragma unroll
    for (int nt = 0; nt < 2; ++nt)
#pragma unroll
      for (int g = 0; g < 8; ++g) {
        int mrow = mb + wm + mt * 16 + (kh << 3) + g;
        int ncol = nb + wn + nt * 16 + r;
        float val = acc[mt][nt][g] + bias[ncol];
        if (mode == 0) {
          int b = mrow >> 12, n = mrow & 4095;
          int h = ncol >> 6, d = ncol & 63;
          obf0[(((size_t)(b * 8 + h) * 4096 + n) << 6) + d] = f2bf(val);
        } else if (mode == 1) {
          of32[(size_t)mrow * N + ncol] = val;
        } else {
          int b = mrow >> 10, n = mrow & 1023;
          int which = ncol >> 9, h = (ncol >> 6) & 7, d = ncol & 63;
          if (which == 0)
            obf0[(((size_t)(b * 8 + h) * 1024 + n) << 6) + d] = f2bf(val);
          else
            obf1[(((size_t)(b * 8 + h) * 64 + d) << 10) + n] = f2bf(val);
        }
      }
}

// ---------------- LayerNorm over C=512, write bf16 ----------------
__global__ __launch_bounds__(TB) void ln_kernel(const float* __restrict__ src,
                                                const float* __restrict__ gamma,
                                                const float* __restrict__ beta,
                                                unsigned short* __restrict__ dst) {
  __shared__ float red[TB];
  const int row = blockIdx.x, tid = threadIdx.x;
  const float* s = src + (size_t)row * 512;
  float v0 = s[tid], v1 = s[tid + 256];
  red[tid] = v0 + v1;
  for (int off = 128; off > 0; off >>= 1) {
    __syncthreads();
    if (tid < off) red[tid] += red[tid + off];
  }
  __syncthreads();
  float mu = red[0] * (1.0f / 512.0f);
  __syncthreads();
  float d0 = v0 - mu, d1 = v1 - mu;
  red[tid] = d0 * d0 + d1 * d1;
  for (int off = 128; off > 0; off >>= 1) {
    __syncthreads();
    if (tid < off) red[tid] += red[tid + off];
  }
  __syncthreads();
  float rstd = rsqrtf(red[0] * (1.0f / 512.0f) + 1e-3f);
  dst[(size_t)row * 512 + tid]       = f2bf(d0 * rstd * gamma[tid] + beta[tid]);
  dst[(size_t)row * 512 + tid + 256] = f2bf(d1 * rstd * gamma[tid + 256] + beta[tid + 256]);
}

// ---------------- fused flash attention ----------------
// q: [B,H,4096,64] bf16 ; k: [B,H,1024,64] bf16 ; vt: [B,H,64,1024] bf16
// ao: [B,4096,512] bf16 (head-major channels, ready for out-proj GEMM)
// block = 256 threads = 8 waves, each wave owns 16 query rows; grid = B*H*32
__global__ __launch_bounds__(TB) void attn_kernel(const unsigned short* __restrict__ q,
                                                  const unsigned short* __restrict__ k,
                                                  const unsigned short* __restrict__ vt,
                                                  unsigned short* __restrict__ ao,
                                                  float scale) {
  __shared__ __align__(16) unsigned short Ks[64 * 64];    // [kv_local][d]
  __shared__ __align__(16) unsigned short Vs[64 * 64];    // [d][kv_local]
  __shared__ __align__(16) unsigned short Ps[8][16 * 64]; // per-wave P staging
  const int tid = threadIdx.x, lane = tid & 31, wave = tid >> 5;
  const int bh = blockIdx.x >> 5;
  const int qb = (blockIdx.x & 31) * 128;
  const int b = bh >> 3, h = bh & 7;
  const unsigned short* qp = q + ((size_t)bh << 18);  // 4096*64
  const unsigned short* kp = k + ((size_t)bh << 16);  // 1024*64
  const unsigned short* vp = vt + ((size_t)bh << 16); // 64*1024
  const int r = lane & 15, kh = lane >> 4;
  const int qr = qb + wave * 16 + r;

  // staging coordinates: 512 vec8 per 64x64 tile, 2 per thread per tile
  const int rr0 = tid >> 3,          cc0 = (tid & 7) * 8;          // rows 0..31
  const int rr1 = (tid + TB) >> 3,   cc1 = ((tid + TB) & 7) * 8;   // rows 32..63

  // Q fragments (A operand, 16x64 split into two 16x32 frags), kept in VGPRs
  v16bf Qf[2];
#pragma unroll
  for (int f = 0; f < 2; ++f) {
    uint4 lo = *(const uint4*)&qp[(size_t)qr * 64 + f * 32 + kh * 8];
    uint4 hi = *(const uint4*)&qp[(size_t)qr * 64 + f * 32 + 16 + kh * 8];
    Qf[f] = frag16(lo, hi);
  }

  v8f zero = {};
  v8f O[4];
#pragma unroll
  for (int dt = 0; dt < 4; ++dt) O[dt] = zero;
  float mprev[8], lsum[8];
#pragma unroll
  for (int g = 0; g < 8; ++g) { mprev[g] = -1e38f; lsum[g] = 0.0f; }

  for (int chunk = 0; chunk < 16; ++chunk) {
    const int m0 = chunk * 64;
    __syncthreads();
#if HAVE_ASYNC_LDS
    async_cp128(&kp[(size_t)(m0 + rr0) * 64 + cc0], &Ks[rr0 * 64 + cc0]);
    async_cp128(&kp[(size_t)(m0 + rr1) * 64 + cc1], &Ks[rr1 * 64 + cc1]);
    async_cp128(&vp[((size_t)rr0 << 10) + m0 + cc0], &Vs[rr0 * 64 + cc0]);
    async_cp128(&vp[((size_t)rr1 << 10) + m0 + cc1], &Vs[rr1 * 64 + cc1]);
    async_wait0();
#else
    uint4 tk0 = *(const uint4*)&kp[(size_t)(m0 + rr0) * 64 + cc0];
    uint4 tk1 = *(const uint4*)&kp[(size_t)(m0 + rr1) * 64 + cc1];
    uint4 tv0 = *(const uint4*)&vp[((size_t)rr0 << 10) + m0 + cc0];
    uint4 tv1 = *(const uint4*)&vp[((size_t)rr1 << 10) + m0 + cc1];
    *(uint4*)&Ks[rr0 * 64 + cc0] = tk0;
    *(uint4*)&Ks[rr1 * 64 + cc1] = tk1;
    *(uint4*)&Vs[rr0 * 64 + cc0] = tv0;
    *(uint4*)&Vs[rr1 * 64 + cc1] = tv1;
#endif
    __syncthreads();

    // S = (Q K^T) * scale  : four 16x16 tiles covering 64 kv rows
    v8f s[4];
#pragma unroll
    for (int nt = 0; nt < 4; ++nt) {
      s[nt] = zero;
#pragma unroll
      for (int f = 0; f < 2; ++f) {
        int col = nt * 16 + r;
        uint4 lo = *(const uint4*)&Ks[col * 64 + f * 32 + kh * 8];
        uint4 hi = *(const uint4*)&Ks[col * 64 + f * 32 + 16 + kh * 8];
        s[nt] = __builtin_amdgcn_wmma_f32_16x16x32_bf16(
            false, Qf[f], false, frag16(lo, hi), (short)0, s[nt], false, false);
      }
      s[nt] = s[nt] * scale;
    }

    // online softmax: row stats across the 16 lanes of each half
    float mc[8];
#pragma unroll
    for (int g = 0; g < 8; ++g) {
      float m = fmaxf(fmaxf(s[0][g], s[1][g]), fmaxf(s[2][g], s[3][g]));
#pragma unroll
      for (int msk = 1; msk < 16; msk <<= 1) m = fmaxf(m, __shfl_xor(m, msk, 32));
      mc[g] = m;
    }
    float mnew[8], alpha[8];
#pragma unroll
    for (int g = 0; g < 8; ++g) {
      mnew[g] = fmaxf(mprev[g], mc[g]);
      alpha[g] = __expf(mprev[g] - mnew[g]);
    }
#pragma unroll
    for (int nt = 0; nt < 4; ++nt)
#pragma unroll
      for (int g = 0; g < 8; ++g) s[nt][g] = __expf(s[nt][g] - mnew[g]);
#pragma unroll
    for (int g = 0; g < 8; ++g) {
      float t = s[0][g] + s[1][g] + s[2][g] + s[3][g];
#pragma unroll
      for (int msk = 1; msk < 16; msk <<= 1) t += __shfl_xor(t, msk, 32);
      lsum[g] = lsum[g] * alpha[g] + t;
    }
#pragma unroll
    for (int dt = 0; dt < 4; ++dt)
#pragma unroll
      for (int g = 0; g < 8; ++g) O[dt][g] *= alpha[g];

    // stage P (C layout -> row-major [16][64] bf16) in wave-private LDS
#pragma unroll
    for (int nt = 0; nt < 4; ++nt)
#pragma unroll
      for (int g = 0; g < 8; ++g)
        Ps[wave][(g + (kh << 3)) * 64 + nt * 16 + r] = f2bf(s[nt][g]);

    // O += P @ V   (A frags from Ps, B frags are kv-contiguous rows of Vs)
    v16bf Af[2];
#pragma unroll
    for (int f = 0; f < 2; ++f) {
      uint4 lo = *(const uint4*)&Ps[wave][r * 64 + f * 32 + kh * 8];
      uint4 hi = *(const uint4*)&Ps[wave][r * 64 + f * 32 + 16 + kh * 8];
      Af[f] = frag16(lo, hi);
    }
#pragma unroll
    for (int dt = 0; dt < 4; ++dt)
#pragma unroll
      for (int f = 0; f < 2; ++f) {
        int d = dt * 16 + r;
        uint4 lo = *(const uint4*)&Vs[d * 64 + f * 32 + kh * 8];
        uint4 hi = *(const uint4*)&Vs[d * 64 + f * 32 + 16 + kh * 8];
        O[dt] = __builtin_amdgcn_wmma_f32_16x16x32_bf16(
            false, Af[f], false, frag16(lo, hi), (short)0, O[dt], false, false);
      }

#pragma unroll
    for (int g = 0; g < 8; ++g) mprev[g] = mnew[g];
  }

  // normalize and scatter to [B, N, h*64+d] bf16
#pragma unroll
  for (int dt = 0; dt < 4; ++dt)
#pragma unroll
    for (int g = 0; g < 8; ++g) {
      int n = qb + wave * 16 + (kh << 3) + g;
      int c = h * 64 + dt * 16 + r;
      ao[(((size_t)b * 4096 + n) << 9) + c] = f2bf(O[dt][g] / lsum[g]);
    }
}

// ---------------------------------------------------------------------------
extern "C" void kernel_launch(void* const* d_in, const int* in_sizes, int n_in,
                              void* d_out, int out_size, void* d_ws, size_t ws_size,
                              hipStream_t stream) {
  (void)in_sizes; (void)n_in; (void)out_size; (void)ws_size;
  const float* x    = (const float*)d_in[0];
  const float* Wq   = (const float*)d_in[3];
  const float* bq   = (const float*)d_in[4];
  const float* Wkv  = (const float*)d_in[5];
  const float* bkv  = (const float*)d_in[6];
  const float* Wp   = (const float*)d_in[7];
  const float* bp   = (const float*)d_in[8];
  const float* srk  = (const float*)d_in[9];
  const float* srb  = (const float*)d_in[10];
  const float* gam  = (const float*)d_in[11];
  const float* bet  = (const float*)d_in[12];
  float* out = (float*)d_out;

  // workspace carve-up (all 256B aligned)
  char* p = (char*)d_ws;
  auto take = [&](size_t bytes) -> void* {
    void* r = (void*)p;
    p += (bytes + 255) & ~(size_t)255;
    return r;
  };
  unsigned short* xbf   = (unsigned short*)take((size_t)16384 * 512 * 2);
  unsigned short* wqT   = (unsigned short*)take((size_t)512 * 512 * 2);
  unsigned short* wkvT  = (unsigned short*)take((size_t)1024 * 512 * 2);
  unsigned short* wpT   = (unsigned short*)take((size_t)512 * 512 * 2);
  unsigned short* srkT  = (unsigned short*)take((size_t)512 * 2048 * 2);
  unsigned short* xp    = (unsigned short*)take((size_t)4096 * 2048 * 2);
  float*          convf = (float*)take((size_t)4096 * 512 * 4);
  unsigned short* xrbf  = (unsigned short*)take((size_t)4096 * 512 * 2);
  unsigned short* qbf   = (unsigned short*)take((size_t)16384 * 512 * 2);
  unsigned short* kbf   = (unsigned short*)take((size_t)32 * 1024 * 64 * 2);
  unsigned short* vtbf  = (unsigned short*)take((size_t)32 * 64 * 1024 * 2);
  unsigned short* aobf  = (unsigned short*)take((size_t)16384 * 512 * 2);

  // bf16 conversions (weights pre-transposed to [N][K])
  cvt_bf16_kernel<<<(16384 * 512 + TB - 1) / TB, TB, 0, stream>>>(x, xbf, 16384 * 512);
  cvtT_bf16_kernel<<<(512 * 512 + TB - 1) / TB, TB, 0, stream>>>(Wq, wqT, 512, 512);
  cvtT_bf16_kernel<<<(512 * 1024 + TB - 1) / TB, TB, 0, stream>>>(Wkv, wkvT, 512, 1024);
  cvtT_bf16_kernel<<<(512 * 512 + TB - 1) / TB, TB, 0, stream>>>(Wp, wpT, 512, 512);
  cvtT_bf16_kernel<<<(2048 * 512 + TB - 1) / TB, TB, 0, stream>>>(srk, srkT, 2048, 512);
  im2row_kernel<<<4096, TB, 0, stream>>>(x, xp);

  // Q projection: [16384x512]@[512x512] -> q packed [B,H,4096,64]
  gemm_bf16_kernel<<<(16384 / BMT) * (512 / BNT), TB, 0, stream>>>(
      xbf, wqT, bq, 16384, 512, 512, 0, qbf, nullptr, nullptr);
  // SR conv as GEMM: [4096x2048]@[2048x512] + srb -> f32
  gemm_bf16_kernel<<<(4096 / BMT) * (512 / BNT), TB, 0, stream>>>(
      xp, srkT, srb, 4096, 512, 2048, 1, nullptr, nullptr, convf);
  // LayerNorm -> bf16
  ln_kernel<<<4096, TB, 0, stream>>>(convf, gam, bet, xrbf);
  // KV projection: [4096x512]@[512x1024] -> k [B,H,1024,64], v^T [B,H,64,1024]
  gemm_bf16_kernel<<<(4096 / BMT) * (1024 / BNT), TB, 0, stream>>>(
      xrbf, wkvT, bkv, 4096, 1024, 512, 2, kbf, vtbf, nullptr);
  // fused attention (scale = 64^-0.5 = 0.125)
  attn_kernel<<<4 * 8 * 32, TB, 0, stream>>>(qbf, kbf, vtbf, aobf, 0.125f);
  // output projection: [16384x512]@[512x512] + bp -> f32 d_out
  gemm_bf16_kernel<<<(16384 / BMT) * (512 / BNT), TB, 0, stream>>>(
      aobf, wpT, bp, 16384, 512, 512, 1, nullptr, nullptr, out);
}
